// dvmsr_net_66271345377413
// MI455X (gfx1250) — matrix operand order repaired and checked by reading, not compiled, not run on hardware
//
#include <hip/hip_runtime.h>
#include <hip/hip_bf16.h>

// ---------------- problem constants ----------------
#define BATCH   2
#define HI      128
#define WI      128
#define LSEQ    (HI * WI)          // 16384
#define MTOT    (BATCH * LSEQ)     // 32768
#define DM      60                 // d_model
#define DI      120                // d_inner
#define DST     16                 // d_state
#define DCONV   4
#define DTR     4
#define XZN     (2 * DI)           // 240
#define XZN_PAD 256                // in_proj N padded to 16 tiles
#define XDBN    (DTR + 2 * DST)    // 36
#define XDB_LD  48                 // padded ld for xdb
#define KP_DM   64                 // d_model K padded
#define KP_DI   128                // d_inner K padded
#define CH      128                // scan chunks
#define CLEN    (LSEQ / CH)        // 128 steps per chunk
#define PFD     8                  // scan prefetch distance

// ---------------- WMMA types ----------------
typedef __attribute__((ext_vector_type(16))) __bf16        v16bf;
typedef __attribute__((ext_vector_type(8)))  float         v8f;
typedef __attribute__((ext_vector_type(4)))  unsigned int  uv4;
typedef int v4i __attribute__((vector_size(4 * sizeof(int))));

union Frag16 { v16bf v; uv4 q[2]; __bf16 e[16]; };

// ---------------- gfx1250 async-load-to-LDS availability ----------------
#if defined(__has_builtin)
#  if __has_builtin(__builtin_amdgcn_global_load_async_to_lds_b128)
#    define HAVE_ASYNC_LDS 1
#  else
#    define HAVE_ASYNC_LDS 0
#  endif
#else
#  define HAVE_ASYNC_LDS 0
#endif

__device__ __forceinline__ void async_stage_16B(const __bf16* src, __bf16* dst)
{
#if HAVE_ASYNC_LDS
    __builtin_amdgcn_global_load_async_to_lds_b128(
        (__attribute__((address_space(1))) v4i*)(src),
        (__attribute__((address_space(3))) v4i*)(dst), 0, 0);
#else
    *(uv4*)dst = *(const uv4*)src;
#endif
}

__device__ __forceinline__ void async_stage_wait()
{
#if HAVE_ASYNC_LDS
#  if defined(__has_builtin) && __has_builtin(__builtin_amdgcn_s_wait_asynccnt)
    __builtin_amdgcn_s_wait_asynccnt(0);
#  else
    asm volatile("s_wait_asynccnt 0x0" ::: "memory");
#  endif
#endif
}

// =====================================================================
// weight prep: fp32 -> bf16, pad K, B-matrix layout [N][Kpad]; A = -exp(Alog)
// =====================================================================
__global__ void prep_weights_k(const float* __restrict__ inw,   // (2,240,60)
                               const float* __restrict__ xpw,   // (2,36,120)
                               const float* __restrict__ outw,  // (2,60,120)
                               const float* __restrict__ alog,  // (2,120,16)
                               __bf16* __restrict__ winbf,      // [2][256][64]
                               __bf16* __restrict__ wxpbf,      // [2][48][128]
                               __bf16* __restrict__ woutbf,     // [2][64][128]
                               float*  __restrict__ aneg)       // [2][120][16]
{
    int i = blockIdx.x * blockDim.x + threadIdx.x;
    int stride = gridDim.x * blockDim.x;
    for (int j = i; j < 2 * XZN_PAD * KP_DM; j += stride) {
        int k = j & (KP_DM - 1); int n = (j >> 6) & (XZN_PAD - 1); int l = j / (XZN_PAD * KP_DM);
        winbf[j] = (__bf16)((k < DM && n < XZN) ? inw[(l * XZN + n) * DM + k] : 0.f);
    }
    for (int j = i; j < 2 * XDB_LD * KP_DI; j += stride) {
        int k = j & (KP_DI - 1); int n = (j >> 7) % XDB_LD; int l = j / (XDB_LD * KP_DI);
        wxpbf[j] = (__bf16)((k < DI && n < XDBN) ? xpw[(l * XDBN + n) * DI + k] : 0.f);
    }
    for (int j = i; j < 2 * 64 * KP_DI; j += stride) {
        int k = j & (KP_DI - 1); int n = (j >> 7) & 63; int l = j / (64 * KP_DI);
        woutbf[j] = (__bf16)((k < DI && n < DM) ? outw[(l * DM + n) * DI + k] : 0.f);
    }
    for (int j = i; j < 2 * DI * DST; j += stride)
        aneg[j] = -__expf(alog[j]);
}

// =====================================================================
// LayerNorm (d_model=60) -> bf16 A-matrix rows [m][64] (pad zeroed)
// one wave per token, wave32 shfl reduction
// =====================================================================
__global__ void layernorm_bf16_k(const float* __restrict__ hin, int ldin,
                                 const float* __restrict__ w,
                                 const float* __restrict__ b,
                                 __bf16* __restrict__ xout)
{
    int lane = threadIdx.x & 31;
    int m = blockIdx.x * (blockDim.x >> 5) + (threadIdx.x >> 5);
    const float* row = hin + (size_t)m * ldin;
    float v0 = row[lane];                                  // lane < 32 < 60
    float v1 = (lane + 32 < DM) ? row[lane + 32] : 0.f;
    float s  = v0 + v1;
    float s2 = v0 * v0 + v1 * v1;
    #pragma unroll
    for (int msk = 16; msk; msk >>= 1) {
        s  += __shfl_xor(s,  msk, 32);
        s2 += __shfl_xor(s2, msk, 32);
    }
    float mu  = s * (1.f / DM);
    float var = s2 * (1.f / DM) - mu * mu;
    float rs  = rsqrtf(var + 1e-5f);
    __bf16* orow = xout + (size_t)m * KP_DM;
    orow[lane] = (__bf16)((v0 - mu) * rs * w[lane] + b[lane]);
    int k1 = lane + 32;
    if (k1 < KP_DM)
        orow[k1] = (k1 < DM) ? (__bf16)((v1 - mu) * rs * w[k1] + b[k1]) : (__bf16)0.f;
}

// =====================================================================
// bf16 WMMA GEMM, B panel staged in LDS via async-load-to-LDS.
// C[M x Nstore] = A[M x Kpad] * B^T  (B stored [Npad][Kpad] row-major)
// block (32,4): 4 M-tiles; each wave computes NT N-tiles (A frag reused).
// grid.y selects the NT-tile group.
// =====================================================================
template<int KPAD, int NT>
__global__ void gemm_bf16_wmma_k(const __bf16* __restrict__ A,
                                 const __bf16* __restrict__ Bw,
                                 float* __restrict__ C,
                                 int Nstore, int ldc)
{
    __shared__ __align__(16) __bf16 Bs[NT * 16 * KPAD];

    const int n0 = blockIdx.y * (NT * 16);
    const __bf16* Bblk = Bw + (size_t)n0 * KPAD;

    // cooperative async stage of the B panel (16B per issue)
    int tid = threadIdx.y * 32 + threadIdx.x;            // 0..127
    constexpr int VECS = NT * 16 * KPAD / 8;             // 16B chunks
    #pragma unroll 4
    for (int i = tid; i < VECS; i += 128)
        async_stage_16B(Bblk + i * 8, Bs + i * 8);
    async_stage_wait();
    __syncthreads();

    const int lane = threadIdx.x;
    const int lh = lane >> 4;
    const int nl = lane & 15;
    const int mtile = blockIdx.x * blockDim.y + threadIdx.y;
    const __bf16* Arow = A + (size_t)(mtile * 16 + nl) * KPAD + lh * 8;

    v8f acc[NT] = {};
    #pragma unroll
    for (int kb = 0; kb < KPAD; kb += 32) {
        Frag16 fa;
        fa.q[0] = *(const uv4*)(Arow + kb);        // K = kb + lh*8 + 0..7
        fa.q[1] = *(const uv4*)(Arow + kb + 16);   // K = kb + 16 + lh*8 + 0..7
        #pragma unroll
        for (int t = 0; t < NT; ++t) {
            const __bf16* Brow = Bs + (t * 16 + nl) * KPAD + kb + lh * 8;
            Frag16 fb;
            fb.q[0] = *(const uv4*)(Brow);
            fb.q[1] = *(const uv4*)(Brow + 16);
            acc[t] = __builtin_amdgcn_wmma_f32_16x16x32_bf16(
                false, fa.v, false, fb.v, (short)0, acc[t], false, false);
        }
    }
    #pragma unroll
    for (int t = 0; t < NT; ++t) {
        int n = n0 + t * 16 + nl;
        if (n < Nstore) {
            float* crow = C + (size_t)(mtile * 16 + lh * 8) * ldc + n;
            #pragma unroll
            for (int r = 0; r < 8; ++r)
                crow[(size_t)r * ldc] = acc[t][r];
        }
    }
}

// =====================================================================
// causal depthwise conv1d (k=4) + SiLU; writes fp32 and bf16(A-matrix) copies
// =====================================================================
__global__ void conv1d_silu_k(const float* __restrict__ xz,   // [m][240] (cols 0..119 = xin)
                              const float* __restrict__ cw,   // [120][4]
                              const float* __restrict__ cb,   // [120]
                              float* __restrict__ xcf,        // [m][120]
                              __bf16* __restrict__ xcbf)      // [m][128]
{
    int idx = blockIdx.x * blockDim.x + threadIdx.x;      // < MTOT*DI
    int c = idx % DI;
    int m = idx / DI;
    int b = m >> 14;            // /LSEQ
    int l = m & (LSEQ - 1);
    float s = cb[c];
    #pragma unroll
    for (int t = 0; t < DCONV; ++t) {
        int lt = l - (DCONV - 1) + t;
        if (lt >= 0)
            s += cw[c * DCONV + t] * xz[(size_t)((b << 14) + lt) * XZN + c];
    }
    float sil = s / (1.f + __expf(-s));
    xcf [(size_t)m * DI    + c] = sil;
    xcbf[(size_t)m * KP_DI + c] = (__bf16)sil;
}

// =====================================================================
// delta = softplus(dt @ dtw^T + dtb)   (DT_RANK = 4, done scalar)
// =====================================================================
__global__ void delta_k(const float* __restrict__ xdb,   // [m][48], cols 0..3 = dt
                        const float* __restrict__ dtw,   // [120][4]
                        const float* __restrict__ dtb,   // [120]
                        float* __restrict__ deltaB)      // [m][120]
{
    int idx = blockIdx.x * blockDim.x + threadIdx.x;
    int d = idx % DI;
    int m = idx / DI;
    const float* row = xdb + (size_t)m * XDB_LD;
    float s = dtb[d];
    #pragma unroll
    for (int r = 0; r < DTR; ++r)
        s += row[r] * dtw[d * DTR + r];
    float dl = (s > 20.f) ? s : log1pf(__expf(s));
    deltaB[(size_t)m * DI + d] = dl;
}

// =====================================================================
// selective scan, pass A: per (b,d,chunk) partial (h with h0=0) + sum(delta)
// 16 lanes = 16 states; unit u = (b*DI+d)*CH + chunk
// =====================================================================
__global__ void scanA_k(const float* __restrict__ deltaB,
                        const float* __restrict__ xdb,
                        const float* __restrict__ xcf,
                        const float* __restrict__ aneg,  // [120][16]
                        float* __restrict__ sumd,        // [units]
                        float* __restrict__ accB)        // [units][16]
{
    int tid = blockIdx.x * blockDim.x + threadIdx.x;
    int n = tid & 15;
    int u = tid >> 4;                       // < 2*120*128
    int chunk = u & (CH - 1);
    int d = (u >> 7) % DI;
    int b = u / (CH * DI);
    float An = aneg[d * DST + n];
    size_t mbase = ((size_t)b << 14) + (size_t)chunk * CLEN;
    float acc = 0.f, sd = 0.f;
    for (int t = 0; t < CLEN; ++t) {
        size_t m = mbase + t;
        if (t + PFD < CLEN) {
            __builtin_prefetch(&deltaB[(m + PFD) * DI + d], 0, 1);
            __builtin_prefetch(&xdb[(m + PFD) * XDB_LD + DTR + n], 0, 1);
            __builtin_prefetch(&xcf[(m + PFD) * DI + d], 0, 1);
        }
        float delta = deltaB[m * DI + d];
        float Bn = xdb[m * XDB_LD + DTR + n];
        float xv = xcf[m * DI + d];
        acc = __expf(delta * An) * acc + delta * Bn * xv;
        sd += delta;
    }
    accB[(size_t)u * DST + n] = acc;
    if (n == 0) sumd[u] = sd;
}

// =====================================================================
// pass B: sequential combine over chunks (240 units x 16 lanes, 128 iters)
// h0B[u,chunk] = state at chunk start; prodA(chunk) = exp(A_n * sum_delta)
// =====================================================================
__global__ void scanB_k(const float* __restrict__ sumd,
                        const float* __restrict__ accB,
                        const float* __restrict__ aneg,
                        float* __restrict__ h0B)
{
    int tid = blockIdx.x * blockDim.x + threadIdx.x;
    int n = tid & 15;
    int u = tid >> 4;                 // b*DI + d
    if (u >= BATCH * DI) return;
    int d = u % DI;
    float An = aneg[d * DST + n];
    float h = 0.f;
    for (int ch = 0; ch < CH; ++ch) {
        size_t uc = (size_t)u * CH + ch;
        h0B[uc * DST + n] = h;
        h = __expf(An * sumd[uc]) * h + accB[uc * DST + n];
    }
}

// =====================================================================
// pass C: rescan chunk from true h0, y = <h,C>; fuse +D*xc and *silu(z);
// emit gated activation as bf16 A-matrix rows [m][128]
// =====================================================================
__global__ void scanC_k(const float* __restrict__ deltaB,
                        const float* __restrict__ xdb,
                        const float* __restrict__ xcf,
                        const float* __restrict__ xz,    // z = cols 120..239
                        const float* __restrict__ aneg,
                        const float* __restrict__ dsk,   // [120]
                        const float* __restrict__ h0B,
                        __bf16* __restrict__ ybf)        // [m][128]
{
    int tid = blockIdx.x * blockDim.x + threadIdx.x;
    int n = tid & 15;
    int u = tid >> 4;
    int chunk = u & (CH - 1);
    int d = (u >> 7) % DI;
    int b = u / (CH * DI);
    float An = aneg[d * DST + n];
    float Dv = dsk[d];
    size_t mbase = ((size_t)b << 14) + (size_t)chunk * CLEN;
    float h = h0B[(size_t)u * DST + n];
    for (int t = 0; t < CLEN; ++t) {
        size_t m = mbase + t;
        if (t + PFD < CLEN) {
            __builtin_prefetch(&deltaB[(m + PFD) * DI + d], 0, 1);
            __builtin_prefetch(&xdb[(m + PFD) * XDB_LD + DTR + n], 0, 1);
            __builtin_prefetch(&xcf[(m + PFD) * DI + d], 0, 1);
        }
        float delta = deltaB[m * DI + d];
        float Bn = xdb[m * XDB_LD + DTR + n];
        float xv = xcf[m * DI + d];
        h = __expf(delta * An) * h + delta * Bn * xv;
        float Cn = xdb[m * XDB_LD + DTR + DST + n];
        float pv = h * Cn;
        pv += __shfl_xor(pv, 1, 32);
        pv += __shfl_xor(pv, 2, 32);
        pv += __shfl_xor(pv, 4, 32);
        pv += __shfl_xor(pv, 8, 32);          // 16-lane group sum
        if (n == 0) {
            float zv = xz[m * XZN + DI + d];
            float yg = (pv + Dv * xv) * (zv / (1.f + __expf(-zv)));
            ybf[m * KP_DI + d] = (__bf16)yg;
        }
    }
}

// =====================================================================
// conv2d 3x3 SAME (60->60) + bias + residual; 9x64 input patch staged in LDS
// block = 64 threads (one output pixel, thread = d_out)
// =====================================================================
__global__ void conv2d_res_k(const float* __restrict__ hbuf,  // [m][64]
                             const float* __restrict__ w,     // [60][60][3][3]
                             const float* __restrict__ bias,  // [60]
                             const float* __restrict__ xorig, // [m][60]
                             float* __restrict__ out)         // [m][60]
{
    __shared__ float tile[9 * 64];
    int pix = blockIdx.x;
    int b = pix / (HI * WI);
    int yy = (pix / WI) % HI;
    int xx = pix % WI;
    int t = threadIdx.x;
    for (int i = t; i < 9 * 64; i += 64) {
        int tap = i >> 6;
        int din = i & 63;
        int y2 = yy + tap / 3 - 1;
        int x2 = xx + tap % 3 - 1;
        float v = 0.f;
        if ((unsigned)y2 < HI && (unsigned)x2 < WI)
            v = hbuf[((size_t)b * LSEQ + (size_t)y2 * WI + x2) * KP_DM + din];
        tile[tap * 64 + din] = v;
    }
    __syncthreads();
    if (t < DM) {
        float acc = bias[t];
        const float* wd = w + (size_t)t * DM * 9;
        for (int din = 0; din < DM; ++din) {
            #pragma unroll
            for (int tap = 0; tap < 9; ++tap)
                acc += wd[din * 9 + tap] * tile[tap * 64 + din];
        }
        size_t m = (size_t)b * LSEQ + (size_t)yy * WI + xx;
        out[m * DM + t] = acc + xorig[m * DM + t];
    }
}

// =====================================================================
// host launch
// =====================================================================
extern "C" void kernel_launch(void* const* d_in, const int* in_sizes, int n_in,
                              void* d_out, int out_size, void* d_ws, size_t ws_size,
                              hipStream_t stream)
{
    const float* x     = (const float*)d_in[0];
    const float* ln_w  = (const float*)d_in[1];
    const float* ln_b  = (const float*)d_in[2];
    const float* in_w  = (const float*)d_in[3];
    const float* cw    = (const float*)d_in[4];
    const float* cb    = (const float*)d_in[5];
    const float* xp_w  = (const float*)d_in[6];
    const float* dt_w  = (const float*)d_in[7];
    const float* dt_b  = (const float*)d_in[8];
    const float* a_log = (const float*)d_in[9];
    const float* d_sk  = (const float*)d_in[10];
    const float* out_w = (const float*)d_in[11];
    const float* c2w   = (const float*)d_in[12];
    const float* c2b   = (const float*)d_in[13];
    float* out = (float*)d_out;

    // ---- workspace layout ----
    char* ws = (char*)d_ws;
    size_t off = 0;
    auto alloc = [&](size_t bytes) -> char* {
        char* p = ws + off;
        off = (off + bytes + 255) & ~(size_t)255;
        return p;
    };
    __bf16* winbf  = (__bf16*)alloc(2 * XZN_PAD * KP_DM * sizeof(__bf16));
    __bf16* wxpbf  = (__bf16*)alloc(2 * XDB_LD * KP_DI * sizeof(__bf16));
    __bf16* woutbf = (__bf16*)alloc(2 * 64 * KP_DI * sizeof(__bf16));
    float*  aneg   = (float*) alloc(2 * DI * DST * sizeof(float));
    __bf16* xln    = (__bf16*)alloc((size_t)MTOT * KP_DM * sizeof(__bf16));
    float*  xz     = (float*) alloc((size_t)MTOT * XZN * sizeof(float));
    float*  xcf    = (float*) alloc((size_t)MTOT * DI * sizeof(float));
    __bf16* xcbf   = (__bf16*)alloc((size_t)MTOT * KP_DI * sizeof(__bf16));
    float*  xdb    = (float*) alloc((size_t)MTOT * XDB_LD * sizeof(float));
    float*  deltaB = (float*) alloc((size_t)MTOT * DI * sizeof(float));
    const int UNITS = BATCH * DI * CH;   // 30720
    float*  sumd   = (float*) alloc((size_t)UNITS * sizeof(float));
    float*  accB   = (float*) alloc((size_t)UNITS * DST * sizeof(float));
    float*  h0B    = (float*) alloc((size_t)UNITS * DST * sizeof(float));
    __bf16* ybf    = (__bf16*)alloc((size_t)MTOT * KP_DI * sizeof(__bf16));
    float*  hout   = (float*) alloc((size_t)MTOT * KP_DM * sizeof(float));

    // zero pad columns of bf16 A-matrices (pads never rewritten afterwards)
    (void)hipMemsetAsync(xcbf, 0, (size_t)MTOT * KP_DI * sizeof(__bf16), stream);
    (void)hipMemsetAsync(ybf,  0, (size_t)MTOT * KP_DI * sizeof(__bf16), stream);

    prep_weights_k<<<64, 256, 0, stream>>>(in_w, xp_w, out_w, a_log,
                                           winbf, wxpbf, woutbf, aneg);

    const float* hin = x;
    int ldin = DM;
    const dim3 gblk(32, 4);

    for (int layer = 0; layer < 2; ++layer) {
        layernorm_bf16_k<<<MTOT / 8, 256, 0, stream>>>(
            hin, ldin, ln_w + layer * DM, ln_b + layer * DM, xln);

        // in_proj: [32768 x 240] = xln[32768 x 64] * winbf^T  (16 tiles, 2 groups of 8)
        gemm_bf16_wmma_k<KP_DM, 8><<<dim3(MTOT / 64, 2), gblk, 0, stream>>>(
            xln, winbf + (size_t)layer * XZN_PAD * KP_DM, xz, XZN, XZN);

        conv1d_silu_k<<<(MTOT * DI) / 256, 256, 0, stream>>>(
            xz, cw + layer * DI * DCONV, cb + layer * DI, xcf, xcbf);

        // x_proj: [32768 x 36] = xcbf[32768 x 128] * wxpbf^T
        gemm_bf16_wmma_k<KP_DI, 3><<<dim3(MTOT / 64, 1), gblk, 0, stream>>>(
            xcbf, wxpbf + (size_t)layer * XDB_LD * KP_DI, xdb, XDBN, XDB_LD);

        delta_k<<<(MTOT * DI) / 256, 256, 0, stream>>>(
            xdb, dt_w + layer * DI * DTR, dt_b + layer * DI, deltaB);

        const float* an = aneg + (size_t)layer * DI * DST;
        scanA_k<<<(UNITS * DST) / 256, 256, 0, stream>>>(deltaB, xdb, xcf, an, sumd, accB);
        scanB_k<<<(BATCH * DI * DST + 255) / 256, 256, 0, stream>>>(sumd, accB, an, h0B);
        scanC_k<<<(UNITS * DST) / 256, 256, 0, stream>>>(
            deltaB, xdb, xcf, xz, an, d_sk + layer * DI, h0B, ybf);

        // out_proj: [32768 x 60(pad 64)] = ybf[32768 x 128] * woutbf^T
        gemm_bf16_wmma_k<KP_DI, 4><<<dim3(MTOT / 64, 1), gblk, 0, stream>>>(
            ybf, woutbf + (size_t)layer * 64 * KP_DI, hout, KP_DM, KP_DM);

        hin = hout;
        ldin = KP_DM;
    }

    conv2d_res_k<<<MTOT, 64, 0, stream>>>(hout, c2w, c2b, x, out);
    (void)in_sizes; (void)n_in; (void)out_size; (void)ws_size;
}
